// LSTMAgent_42511586296163
// MI455X (gfx1250) — matrix operand order repaired and checked by reading, not compiled, not run on hardware
//
#include <hip/hip_runtime.h>
#include <math.h>

// ---------------------------------------------------------------------------
// Types for CDNA5 WMMA (wave32): A/B = v16h (16 x f16 = 8 VGPRs), C/D = v8f
// ---------------------------------------------------------------------------
typedef _Float16 h16;
typedef _Float16 v16h __attribute__((ext_vector_type(16)));
typedef _Float16 v8h  __attribute__((ext_vector_type(8)));
typedef float    v8f  __attribute__((ext_vector_type(8)));

#define T_STEPS 512
#define BATCH   256
#define HID     512
#define GATES   2048          // 4*HID
#define OBS     128
#define FEATD   256
#define TB      131072        // T_STEPS * BATCH
#define NBLK    64            // persistent-kernel workgroups (512 waves)
#define KTOT    768           // FEATD + HID
#define LDS_STRIDE 776        // 768 + 8 halves of pad (bank stagger)

__device__ __forceinline__ v8f wmma_f16(v16h a, v16h b, v8f c) {
  return __builtin_amdgcn_wmma_f32_16x16x32_f16(
      false, a, false, b, (short)0, c, false, false);
}

// A operand: row-major M x K, tile origin (r0,k0).
// ISA 16-bit A 16x32: lanes 0-15 (row m): K 0-7 + K 16-23; lanes 16-31: K 8-15 + K 24-31.
__device__ __forceinline__ v16h load_fragA(const h16* __restrict__ base, int ld,
                                           int r0, int k0) {
  const int lane = threadIdx.x & 31;
  const int lo = lane & 15, hi = lane >> 4;
  const h16* p = base + (size_t)(r0 + lo) * ld + (k0 + hi * 8);
  v8h a = *(const v8h*)(p);
  v8h b = *(const v8h*)(p + 16);
  v16h r;
#pragma unroll
  for (int i = 0; i < 8; ++i) { r[i] = a[i]; r[i + 8] = b[i]; }
  return r;
}

// B operand from weights stored row-major N x K (computing A @ W^T):
// ISA 16-bit B 32x16: lane n holds K 0-7 + K 16-23 of col n; lane n+16 holds K 8-15 + K 24-31.
__device__ __forceinline__ v16h load_fragB(const h16* __restrict__ W, int ld,
                                           int n0, int k0) {
  const int lane = threadIdx.x & 31;
  const int lo = lane & 15, hi = lane >> 4;
  const h16* p = W + (size_t)(n0 + lo) * ld + (k0 + hi * 8);
  v8h a = *(const v8h*)(p);
  v8h b = *(const v8h*)(p + 16);
  v16h r;
#pragma unroll
  for (int i = 0; i < 8; ++i) { r[i] = a[i]; r[i + 8] = b[i]; }
  return r;
}

__device__ __forceinline__ float sigm(float x) {
  return 1.0f / (1.0f + __expf(-x));
}

// CDNA5 async global->LDS copy, 16B per lane. Tracked by ASYNCcnt.
__device__ __forceinline__ void async_g2l_b128(unsigned lds_off, const h16* gaddr) {
  asm volatile("global_load_async_to_lds_b128 %0, %1, off"
               :: "v"(lds_off), "v"(gaddr) : "memory");
}
__device__ __forceinline__ void wait_asynccnt0() {
  asm volatile("s_wait_asynccnt 0x0" ::: "memory");
}

// ---------------------------------------------------------------------------
// Small prep kernels
// ---------------------------------------------------------------------------
__global__ void k_f32_to_f16(const float* __restrict__ in, h16* __restrict__ out, int n) {
  for (int i = blockIdx.x * blockDim.x + threadIdx.x; i < n; i += gridDim.x * blockDim.x)
    out[i] = (h16)in[i];
}

__global__ void k_pack_head(const float* __restrict__ Wa, const float* __restrict__ Wc,
                            h16* __restrict__ Wh) {
  int i = blockIdx.x * blockDim.x + threadIdx.x;
  if (i >= 16 * HID) return;
  int r = i >> 9, k = i & (HID - 1);
  float v = (r < 8) ? Wa[r * HID + k] : ((r == 8) ? Wc[k] : 0.0f);
  Wh[i] = (h16)v;
}

__global__ void k_bias_gates(const float* __restrict__ bih, const float* __restrict__ bhh,
                             float* __restrict__ bg) {
  int i = blockIdx.x * blockDim.x + threadIdx.x;
  if (i < GATES) bg[i] = bih[i] + bhh[i];
}

__global__ void k_init_state(const float* __restrict__ h0, const float* __restrict__ c0,
                             const int* __restrict__ done, h16* __restrict__ hm,
                             float* __restrict__ cbuf, unsigned* __restrict__ ctrl) {
  int i = blockIdx.x * blockDim.x + threadIdx.x;
  if (i == 0) ctrl[0] = 0u;
  if (i < BATCH * HID) {
    int m = i >> 9;
    float mk = 1.0f - (float)done[m];
    hm[i]   = (h16)(h0[i] * mk);
    cbuf[i] = c0[i] * mk;
  }
}

// ---------------------------------------------------------------------------
// Generic one-tile-per-wave WMMA GEMM:  out = act(A(MxK,f16) @ W(NxK,f16)^T + b)
// ---------------------------------------------------------------------------
__global__ void __launch_bounds__(256) k_gemm_relu(
    const h16* __restrict__ A, const h16* __restrict__ W,
    const float* __restrict__ bias, h16* __restrict__ out,
    int M, int N, int K, int do_relu) {
  const int gw = blockIdx.x * (blockDim.x >> 5) + (threadIdx.x >> 5);
  const int ntiles = N >> 4;
  const int mt = gw / ntiles, nt = gw - mt * ntiles;
  const int m0 = mt << 4, n0 = nt << 4;
  if (m0 >= M) return;
  v8f acc = {0.f, 0.f, 0.f, 0.f, 0.f, 0.f, 0.f, 0.f};
  for (int k = 0; k < K; k += 32)
    acc = wmma_f16(load_fragA(A, K, m0, k), load_fragB(W, K, n0, k), acc);
  const int lane = threadIdx.x & 31;
  const int lo = lane & 15, hi = lane >> 4;
  const float b = bias[n0 + lo];
#pragma unroll
  for (int r = 0; r < 8; ++r) {
    float v = acc[r] + b;
    if (do_relu) v = v > 0.0f ? v : 0.0f;
    out[(size_t)(m0 + r + hi * 8) * N + (n0 + lo)] = (h16)v;
  }
}

// ---------------------------------------------------------------------------
// Device-wide barrier (monotonic counter).
// ---------------------------------------------------------------------------
__device__ __forceinline__ void grid_barrier(unsigned* __restrict__ ctrl, unsigned target) {
  __builtin_amdgcn_fence(__ATOMIC_RELEASE, "agent");
  __syncthreads();
  if (threadIdx.x == 0) {
    __hip_atomic_fetch_add(ctrl, 1u, __ATOMIC_RELAXED, __HIP_MEMORY_SCOPE_AGENT);
    while (__hip_atomic_load(ctrl, __ATOMIC_RELAXED, __HIP_MEMORY_SCOPE_AGENT) < target)
      __builtin_amdgcn_s_sleep(2);
  }
  __syncthreads();
  __builtin_amdgcn_fence(__ATOMIC_ACQUIRE, "agent");
}

// ---------------------------------------------------------------------------
// Persistent LSTM scan. 64 WGs x 256 threads = 512 waves.
// WG b: M-group mg = b>>4 (64 rows of A), N-tiles [ (b&15)*8 , +8 ).
// Per step: WG async-copies its 64 x 768 f16 A-panel (feat_t || h_masked) into
// LDS (97KB, padded rows), then each wave computes a 64x16 gate patch:
// one B fragment from L2 feeds 4 WMMAs (A fragments re-read from LDS).
// ---------------------------------------------------------------------------
__global__ void __launch_bounds__(256, 1) k_lstm_scan(
    const h16* __restrict__ feat, const h16* __restrict__ Wih,
    const h16* __restrict__ Whh, const float* __restrict__ bg,
    const int* __restrict__ done,
    h16* __restrict__ hm, float* __restrict__ cbuf, float* __restrict__ gates,
    h16* __restrict__ hs, float* __restrict__ out_hn, float* __restrict__ out_cn,
    unsigned* __restrict__ ctrl) {
  extern __shared__ h16 smem[];   // 64 * LDS_STRIDE halves = 99,328 B

  const int lane = threadIdx.x & 31;
  const int lo = lane & 15, hi = lane >> 4;
  const int mg = blockIdx.x >> 4;                              // 0..3
  const int m0 = mg << 6;                                      // A row base
  const int nt = ((blockIdx.x & 15) << 3) + (threadIdx.x >> 5); // 0..127
  const int n0 = nt << 4;
  unsigned barNum = 0;

  for (int t = 0; t < T_STEPS; ++t) {
    const h16* ft = feat + (size_t)t * BATCH * FEATD;

    // ---- async fill of the 64x768 A panel: [feat_t rows | hm rows] ----
    // 96 x 16B chunks per row (32 feat + 64 hm), 64 rows = 6144 chunks.
    for (int c = threadIdx.x; c < 64 * 96; c += 256) {
      const int row = c / 96, wi = c - row * 96;
      const h16* src;
      unsigned dst;
      if (wi < 32) {
        src = ft + (size_t)(m0 + row) * FEATD + wi * 8;
        dst = (unsigned)(size_t)&smem[row * LDS_STRIDE + wi * 8];
      } else {
        const int w2 = wi - 32;
        src = hm + (size_t)(m0 + row) * HID + w2 * 8;
        dst = (unsigned)(size_t)&smem[row * LDS_STRIDE + FEATD + w2 * 8];
      }
      async_g2l_b128(dst, src);
    }
    wait_asynccnt0();
    __syncthreads();

    // ---- gates(64x16 patch) = A_panel @ [Wih|Whh]^T + bias ----
    v8f acc0 = {0.f,0.f,0.f,0.f,0.f,0.f,0.f,0.f};
    v8f acc1 = acc0, acc2 = acc0, acc3 = acc0;
    for (int k = 0; k < FEATD; k += 32) {
      const v16h b = load_fragB(Wih, FEATD, n0, k);
      acc0 = wmma_f16(load_fragA(smem, LDS_STRIDE,  0, k), b, acc0);
      acc1 = wmma_f16(load_fragA(smem, LDS_STRIDE, 16, k), b, acc1);
      acc2 = wmma_f16(load_fragA(smem, LDS_STRIDE, 32, k), b, acc2);
      acc3 = wmma_f16(load_fragA(smem, LDS_STRIDE, 48, k), b, acc3);
    }
    for (int k = 0; k < HID; k += 32) {
      const v16h b = load_fragB(Whh, HID, n0, k);
      acc0 = wmma_f16(load_fragA(smem, LDS_STRIDE,  0, FEATD + k), b, acc0);
      acc1 = wmma_f16(load_fragA(smem, LDS_STRIDE, 16, FEATD + k), b, acc1);
      acc2 = wmma_f16(load_fragA(smem, LDS_STRIDE, 32, FEATD + k), b, acc2);
      acc3 = wmma_f16(load_fragA(smem, LDS_STRIDE, 48, FEATD + k), b, acc3);
    }
    const float bb = bg[n0 + lo];
#pragma unroll
    for (int r = 0; r < 8; ++r) {
      const int rr = r + hi * 8;
      gates[(size_t)(m0 +  0 + rr) * GATES + (n0 + lo)] = acc0[r] + bb;
      gates[(size_t)(m0 + 16 + rr) * GATES + (n0 + lo)] = acc1[r] + bb;
      gates[(size_t)(m0 + 32 + rr) * GATES + (n0 + lo)] = acc2[r] + bb;
      gates[(size_t)(m0 + 48 + rr) * GATES + (n0 + lo)] = acc3[r] + bb;
    }
    grid_barrier(ctrl, ++barNum * NBLK);

    // ---- cell update (i,f,g,o); apply mask[t+1] for next step ----
    for (int e = blockIdx.x * blockDim.x + threadIdx.x; e < BATCH * HID;
         e += NBLK * 256) {
      const int m = e >> 9, j = e & (HID - 1);
      const size_t gb = (size_t)m * GATES + j;
      const float ig = gates[gb];
      const float fg = gates[gb + 512];
      const float gg = gates[gb + 1024];
      const float og = gates[gb + 1536];
      const float c = sigm(fg) * cbuf[e] + sigm(ig) * tanhf(gg);
      const float h = sigm(og) * tanhf(c);
      hs[(size_t)t * (BATCH * HID) + e] = (h16)h;
      if (t < T_STEPS - 1) {
        const float mk = 1.0f - (float)done[(t + 1) * BATCH + m];
        hm[e]   = (h16)(h * mk);
        cbuf[e] = c * mk;
      } else {
        out_hn[e] = h;
        out_cn[e] = c;
      }
    }
    grid_barrier(ctrl, ++barNum * NBLK);
  }
}

// ---------------------------------------------------------------------------
// Output heads: [logits | value] = hs(TBx512,f16) @ Whead(16x512,f16)^T + b
// ---------------------------------------------------------------------------
__global__ void __launch_bounds__(256) k_heads(
    const h16* __restrict__ hs, const h16* __restrict__ Wh,
    const float* __restrict__ ba, const float* __restrict__ bc,
    float* __restrict__ logits, float* __restrict__ value) {
  const int gw = blockIdx.x * (blockDim.x >> 5) + (threadIdx.x >> 5);
  const int m0 = gw << 4;
  v8f acc = {0.f, 0.f, 0.f, 0.f, 0.f, 0.f, 0.f, 0.f};
  for (int k = 0; k < HID; k += 32)
    acc = wmma_f16(load_fragA(hs, HID, m0, k), load_fragB(Wh, HID, 0, k), acc);
  const int lane = threadIdx.x & 31;
  const int lo = lane & 15, hi = lane >> 4;
#pragma unroll
  for (int r = 0; r < 8; ++r) {
    const int row = m0 + r + hi * 8;
    const float v = acc[r];
    if (lo < 8)       logits[(size_t)row * 8 + lo] = v + ba[lo];
    else if (lo == 8) value[row] = v + bc[0];
  }
}

// ---------------------------------------------------------------------------
extern "C" void kernel_launch(void* const* d_in, const int* in_sizes, int n_in,
                              void* d_out, int out_size, void* d_ws, size_t ws_size,
                              hipStream_t stream) {
  (void)in_sizes; (void)n_in; (void)out_size; (void)ws_size;
  const float* x    = (const float*)d_in[0];
  const int*   done = (const int*)  d_in[1];
  const float* h0   = (const float*)d_in[2];
  const float* c0   = (const float*)d_in[3];
  const float* W1   = (const float*)d_in[4];
  const float* b1   = (const float*)d_in[5];
  const float* W2   = (const float*)d_in[6];
  const float* b2   = (const float*)d_in[7];
  const float* W3   = (const float*)d_in[8];
  const float* b3   = (const float*)d_in[9];
  const float* Wih  = (const float*)d_in[10];
  const float* Whh  = (const float*)d_in[11];
  const float* bih  = (const float*)d_in[12];
  const float* bhh  = (const float*)d_in[13];
  const float* Wa   = (const float*)d_in[14];
  const float* ba   = (const float*)d_in[15];
  const float* Wc   = (const float*)d_in[16];
  const float* bc   = (const float*)d_in[17];

  float* out    = (float*)d_out;
  float* logits = out;                       // TB x 8
  float* value  = out + (size_t)TB * 8;      // TB x 1
  float* out_hn = value + TB;                // B x H
  float* out_cn = out_hn + BATCH * HID;      // B x H

  char* ws = (char*)d_ws;
  size_t off = 0;
  auto carve = [&](size_t bytes) -> char* {
    char* p = ws + off;
    off = (off + bytes + 255) & ~(size_t)255;
    return p;
  };
  unsigned* ctrl = (unsigned*)carve(256);
  h16* xf    = (h16*)carve((size_t)TB * OBS * 2);
  h16* h1f   = (h16*)carve((size_t)TB * 128 * 2);
  h16* h2f   = (h16*)carve((size_t)TB * 32 * 2);
  h16* featf = (h16*)carve((size_t)TB * FEATD * 2);
  h16* hsf   = (h16*)carve((size_t)TB * HID * 2);
  h16* W1h   = (h16*)carve(128 * 128 * 2);
  h16* W2h   = (h16*)carve(32 * 128 * 2);
  h16* W3h   = (h16*)carve(256 * 32 * 2);
  h16* Wihh  = (h16*)carve((size_t)GATES * FEATD * 2);
  h16* Whhh  = (h16*)carve((size_t)GATES * HID * 2);
  h16* Whead = (h16*)carve(16 * HID * 2);
  float* bg  = (float*)carve(GATES * 4);
  h16* hm    = (h16*)carve(BATCH * HID * 2);
  float* cb  = (float*)carve(BATCH * HID * 4);
  float* gts = (float*)carve((size_t)BATCH * GATES * 4);

  auto cvt = [&](const float* src, h16* dst, int n) {
    int blocks = (n + 255) / 256;
    k_f32_to_f16<<<blocks, 256, 0, stream>>>(src, dst, n);
  };
  cvt(x,   xf,   TB * OBS);
  cvt(W1,  W1h,  128 * 128);
  cvt(W2,  W2h,  32 * 128);
  cvt(W3,  W3h,  256 * 32);
  cvt(Wih, Wihh, GATES * FEATD);
  cvt(Whh, Whhh, GATES * HID);
  k_pack_head<<<(16 * HID + 255) / 256, 256, 0, stream>>>(Wa, Wc, Whead);
  k_bias_gates<<<(GATES + 255) / 256, 256, 0, stream>>>(bih, bhh, bg);

  // ---- MLP encoder ----
  k_gemm_relu<<<8192, 256, 0, stream>>>(xf,  W1h, b1, h1f,  TB, 128, 128, 1);
  k_gemm_relu<<<2048, 256, 0, stream>>>(h1f, W2h, b2, h2f,  TB, 32, 128, 1);
  k_gemm_relu<<<16384, 256, 0, stream>>>(h2f, W3h, b3, featf, TB, 256, 32, 1);

  // ---- initial state (mask at t=0) + barrier-counter reset ----
  k_init_state<<<(BATCH * HID + 255) / 256, 256, 0, stream>>>(h0, c0, done, hm, cb, ctrl);

  // ---- persistent recurrent scan (dynamic LDS: 64 x 776 halves) ----
  k_lstm_scan<<<NBLK, 256, (size_t)64 * LDS_STRIDE * 2, stream>>>(
      featf, Wihh, Whhh, bg, done, hm, cb, gts, hsf, out_hn, out_cn, ctrl);

  // ---- heads ----
  k_heads<<<1024, 256, 0, stream>>>(hsf, Whead, ba, bc, logits, value);
}
